// ElasticAttention_48095043780938
// MI455X (gfx1250) — matrix operand order repaired and checked by reading, not compiled, not run on hardware
//
#include <hip/hip_runtime.h>
#include <hip/hip_bf16.h>
#include <math.h>

typedef __attribute__((ext_vector_type(16))) __bf16 v16bf;
typedef __attribute__((ext_vector_type(8)))  float  v8f;
typedef __attribute__((ext_vector_type(4)))  int    v4i;
typedef __attribute__((address_space(1))) v4i gv4i;   // global int4
typedef __attribute__((address_space(3))) v4i lv4i;   // LDS int4

#define EMBED  1024
#define NHEADS 16
#define HDIM   64
#define BATCH  4
#define SEQ    2048
#define MROWS  (BATCH*SEQ)   // 8192

#if __has_builtin(__builtin_amdgcn_global_load_async_to_lds_b128)
#define USE_ASYNC_LDS 1
#else
#define USE_ASYNC_LDS 0
#endif

// Copy 16 bytes global -> LDS. Async (GLOBAL_LOAD_ASYNC_TO_LDS_B128, ASYNCcnt)
// when the toolchain exposes the builtin, else a plain load+store pair.
__device__ __forceinline__ void cp16_to_lds(void* lds, const void* g) {
#if USE_ASYNC_LDS
  // flat numeric value == AS1 value; generic LDS pointer low 32 bits == LDS offset
  __builtin_amdgcn_global_load_async_to_lds_b128(
      (gv4i*)(unsigned long long)g,
      (lv4i*)(unsigned int)(unsigned long long)lds,
      0, 0);
#else
  *(uint4*)lds = *(const uint4*)g;
#endif
}

__device__ __forceinline__ void wait_lds_async() {
#if USE_ASYNC_LDS
#if __has_builtin(__builtin_amdgcn_s_wait_asynccnt)
  __builtin_amdgcn_s_wait_asynccnt(0);
#else
  asm volatile("s_wait_asynccnt 0" ::: "memory");
#endif
#endif
}

__device__ __forceinline__ v8f zero8() {
  v8f z = {0.f,0.f,0.f,0.f,0.f,0.f,0.f,0.f};
  return z;
}

__device__ __forceinline__ v8f wmma_bf16(v16bf a, v16bf b, v8f c) {
  return __builtin_amdgcn_wmma_f32_16x16x32_bf16(false, a, false, b, (short)0, c, false, false);
}

// A-fragment: 16x32 bf16 tile at p (row-major, `stride` elems/row).
// lane m=lane&15 holds row m; K = (e>>3)*16 + hi*8 + (e&7)  -> two b128 loads.
__device__ __forceinline__ v16bf frag_a(const __bf16* p, int stride) {
  const int lane = threadIdx.x & 31;
  const int m = lane & 15, hi = lane >> 4;
  const __bf16* row = p + m * stride + hi * 8;
  v16bf a;
#pragma unroll
  for (int e = 0; e < 16; ++e) {
    int k = ((e >> 3) << 4) + (e & 7);
    a[e] = row[k];
  }
  return a;
}

// B-fragment from an N-major tile (row n, k contiguous): B(k,n) = p[n*stride + k].
// lane n=lane&15 holds col n; K = (e>>3)*16 + hi*8 + (e&7) -> two b128 loads.
__device__ __forceinline__ v16bf frag_b_nk(const __bf16* p, int stride) {
  const int lane = threadIdx.x & 31;
  const int n = lane & 15, hi = lane >> 4;
  const __bf16* row = p + n * stride + hi * 8;
  v16bf b;
#pragma unroll
  for (int e = 0; e < 16; ++e) b[e] = row[((e >> 3) << 4) + (e & 7)];
  return b;
}

// ---------------- Kernel 0: fp32 -> bf16 convert ----------------
__global__ __launch_bounds__(256) void cvt_f32_bf16(const float* __restrict__ x,
                                                    __bf16* __restrict__ y, int n) {
  int i = (blockIdx.x * blockDim.x + threadIdx.x) * 4;
  if (i + 3 < n) {
    float4 v = *(const float4*)(x + i);
    y[i + 0] = (__bf16)v.x;
    y[i + 1] = (__bf16)v.y;
    y[i + 2] = (__bf16)v.z;
    y[i + 3] = (__bf16)v.w;
  }
}

// ---------------- Kernel 1: fused QKV projection GEMM ----------------
// Y[m,f] = sum_k X[m,k]*W[f,k] + bias[f]; W kept in native [f,k] layout (N-major,
// which is exactly the WMMA B-fragment-friendly orientation). Q scaled by D^-0.5.
__global__ __launch_bounds__(256)
void qkv_gemm(const __bf16* __restrict__ X, const __bf16* __restrict__ Wqb,
              const __bf16* __restrict__ Wkb, const __bf16* __restrict__ Wvb,
              const float* __restrict__ bq, const float* __restrict__ bk,
              const float* __restrict__ bv, __bf16* __restrict__ Q,
              __bf16* __restrict__ K, __bf16* __restrict__ V) {
  constexpr int TS = 40;    // LDS row stride for 128x32 tiles (80B: 16B aligned)
  __shared__ __bf16 As[128 * TS];
  __shared__ __bf16 Bs[128 * TS];

  const int mat = blockIdx.z;
  const __bf16* Wb = (mat == 0) ? Wqb : (mat == 1) ? Wkb : Wvb;
  const float* bias = (mat == 0) ? bq : (mat == 1) ? bk : bv;
  __bf16* out = (mat == 0) ? Q : (mat == 1) ? K : V;

  const int bm = blockIdx.x * 128;
  const int bn = blockIdx.y * 128;
  const int t = threadIdx.x;
  const int w = t >> 5, lane = t & 31;
  const int mw = w >> 1, nw = w & 1;       // 4x2 wave grid: 32 rows x 64 cols each
  const int hi = lane >> 4, ln = lane & 15;

  v8f acc[2][4];
#pragma unroll
  for (int i = 0; i < 2; ++i)
#pragma unroll
    for (int j = 0; j < 4; ++j) acc[i][j] = zero8();

  for (int k0 = 0; k0 < EMBED; k0 += 32) {
    __syncthreads();
    // stage A (rows of X) and B (rows of W): each 128x32 bf16 = 512 x 16B
#pragma unroll
    for (int i = 0; i < 2; ++i) {
      int u = t + i * 256;
      int row = u >> 2, seg = u & 3;
      cp16_to_lds(As + row * TS + seg * 8,
                  X + (size_t)(bm + row) * EMBED + k0 + seg * 8);
      cp16_to_lds(Bs + row * TS + seg * 8,
                  Wb + (size_t)(bn + row) * EMBED + k0 + seg * 8);
    }
    wait_lds_async();
    __syncthreads();

    v16bf af[2], bfr[4];
#pragma unroll
    for (int mi = 0; mi < 2; ++mi) af[mi] = frag_a(As + (mw * 32 + mi * 16) * TS, TS);
#pragma unroll
    for (int ni = 0; ni < 4; ++ni) bfr[ni] = frag_b_nk(Bs + (nw * 64 + ni * 16) * TS, TS);
#pragma unroll
    for (int mi = 0; mi < 2; ++mi)
#pragma unroll
      for (int ni = 0; ni < 4; ++ni)
        acc[mi][ni] = wmma_bf16(af[mi], bfr[ni], acc[mi][ni]);
  }

  const float qscale = (mat == 0) ? 0.125f : 1.0f;  // D^-0.5 folded into Q
#pragma unroll
  for (int mi = 0; mi < 2; ++mi)
#pragma unroll
    for (int ni = 0; ni < 4; ++ni)
#pragma unroll
      for (int r = 0; r < 8; ++r) {
        int m = bm + mw * 32 + mi * 16 + r + hi * 8;
        int f = bn + nw * 64 + ni * 16 + ln;
        float v = (acc[mi][ni][r] + bias[f]) * qscale;
        int b = m >> 11, s = m & 2047;
        int h = f >> 6, d = f & 63;
        out[(((size_t)(b * NHEADS + h) * SEQ) + s) * HDIM + d] = (__bf16)v;
      }
}

// ---------------- Kernel 2: flash attention per (b,h,128-row q tile) ----------------
__global__ __launch_bounds__(256)
void flash_attn(const __bf16* __restrict__ Q, const __bf16* __restrict__ K,
                const __bf16* __restrict__ V, __bf16* __restrict__ Attn,
                const int* __restrict__ activep) {
  constexpr int KS = 72;  // 144B row stride: 16B aligned, conflict-free frag reads
  __shared__ __bf16 Kt[64 * KS];    // K tile, [s'][d]
  __shared__ __bf16 VtT[64 * KS];   // V tile transposed, [d][s']
  __shared__ __bf16 Pb[8][16 * KS];

  const int qt = blockIdx.x;   // 0..15 (128-row q tiles)
  const int h  = blockIdx.y;   // 0..15
  const int b  = blockIdx.z;   // 0..3
  const int t = threadIdx.x, w = t >> 5, lane = t & 31;
  const int hi = lane >> 4, ln = lane & 15;
  const int active = activep[0];

  if (h >= active) {
    // inactive head: v is masked -> attention output is exactly zero
    uint4 z = {0u, 0u, 0u, 0u};
#pragma unroll
    for (int i = 0; i < 4; ++i) {
      int u = t + i * 256;          // 1024 x 16B = 128 rows x 64 bf16
      int row = u >> 3, seg = u & 7;
      *((uint4*)(Attn + ((size_t)(b * SEQ + qt * 128 + row) * EMBED) + h * HDIM) + seg) = z;
    }
    return;
  }

  const __bf16* Qg = Q + ((size_t)(b * NHEADS + h) * SEQ + qt * 128 + w * 16) * HDIM;
  const __bf16* Kg = K + ((size_t)(b * NHEADS + h) * SEQ) * HDIM;
  const __bf16* Vg = V + ((size_t)(b * NHEADS + h) * SEQ) * HDIM;

  v16bf qf[2];
#pragma unroll
  for (int kk = 0; kk < 2; ++kk) qf[kk] = frag_a(Qg + kk * 32, HDIM);

  v8f o[4];
#pragma unroll
  for (int i = 0; i < 4; ++i) o[i] = zero8();
  float mrow[8], lrow[8];
#pragma unroll
  for (int r = 0; r < 8; ++r) { mrow[r] = -3.0e38f; lrow[r] = 0.f; }

  __bf16* Pw = &Pb[w][0];

  for (int kt = 0; kt < SEQ; kt += 64) {
    __syncthreads();
    // stage K tile (async path) and V tile transposed (manual, element scatter)
#pragma unroll
    for (int i = 0; i < 2; ++i) {
      int u = t + i * 256;
      int row = u >> 3, seg = u & 7;     // row = s' 0..63, seg*8 = d base
      cp16_to_lds(Kt + row * KS + seg * 8,
                  Kg + (size_t)(kt + row) * HDIM + seg * 8);
      uint4 d4 = *((const uint4*)(Vg + (size_t)(kt + row) * HDIM) + seg);
      __bf16 tmp[8];
      *(uint4*)tmp = d4;
#pragma unroll
      for (int j = 0; j < 8; ++j)
        VtT[(seg * 8 + j) * KS + row] = tmp[j];
    }
    wait_lds_async();
    __syncthreads();

    // scores S = Qs * K^T : 16 x 64 per wave (B-frag contiguous from Kt rows)
    v8f sc[4];
#pragma unroll
    for (int ni = 0; ni < 4; ++ni) sc[ni] = zero8();
#pragma unroll
    for (int ni = 0; ni < 4; ++ni)
#pragma unroll
      for (int kk = 0; kk < 2; ++kk)
        sc[ni] = wmma_bf16(qf[kk], frag_b_nk(Kt + (ni * 16) * KS + kk * 32, KS), sc[ni]);

    // online softmax (row stats per lane for rows r+hi*8; reduce across 16 lanes)
    float mnew[8], alpha[8];
#pragma unroll
    for (int r = 0; r < 8; ++r) {
      float mx = sc[0][r];
#pragma unroll
      for (int ni = 1; ni < 4; ++ni) mx = fmaxf(mx, sc[ni][r]);
#pragma unroll
      for (int off = 1; off < 16; off <<= 1) mx = fmaxf(mx, __shfl_xor(mx, off, 32));
      mnew[r] = fmaxf(mrow[r], mx);
      alpha[r] = __expf(mrow[r] - mnew[r]);
    }
#pragma unroll
    for (int r = 0; r < 8; ++r) {
      float rs = 0.f;
#pragma unroll
      for (int ni = 0; ni < 4; ++ni) {
        float pv = __expf(sc[ni][r] - mnew[r]);
        rs += pv;
        Pw[(r + hi * 8) * KS + ni * 16 + ln] = (__bf16)pv;  // C-layout -> LDS
      }
#pragma unroll
      for (int off = 1; off < 16; off <<= 1) rs += __shfl_xor(rs, off, 32);
      lrow[r] = lrow[r] * alpha[r] + rs;
      mrow[r] = mnew[r];
#pragma unroll
      for (int ni = 0; ni < 4; ++ni) o[ni][r] *= alpha[r];
    }
    __syncthreads();  // order P stores before A-fragment reloads

    // O += P * V  (V fragments contiguous from transposed VtT rows)
#pragma unroll
    for (int kk = 0; kk < 2; ++kk) {
      v16bf pf = frag_a(Pw + kk * 32, KS);
#pragma unroll
      for (int nd = 0; nd < 4; ++nd)
        o[nd] = wmma_bf16(pf, frag_b_nk(VtT + (nd * 16) * KS + kk * 32, KS), o[nd]);
    }
  }

  // normalize and store to Attn [B,S,E] bf16
#pragma unroll
  for (int r = 0; r < 8; ++r) {
    float inv = 1.0f / lrow[r];
    int s = qt * 128 + w * 16 + r + hi * 8;
#pragma unroll
    for (int nd = 0; nd < 4; ++nd) {
      int e = h * HDIM + nd * 16 + ln;
      Attn[(size_t)(b * SEQ + s) * EMBED + e] = (__bf16)(o[nd][r] * inv);
    }
  }
}

// ---------------- Kernel 3: output projection (fp32 epilogue) ----------------
__global__ __launch_bounds__(256)
void out_gemm(const __bf16* __restrict__ A, const __bf16* __restrict__ Wob,
              const float* __restrict__ bo, float* __restrict__ Out) {
  constexpr int TS = 40;
  __shared__ __bf16 As[128 * TS];
  __shared__ __bf16 Bs[128 * TS];

  const int bm = blockIdx.x * 128;
  const int bn = blockIdx.y * 128;
  const int t = threadIdx.x;
  const int w = t >> 5, lane = t & 31;
  const int mw = w >> 1, nw = w & 1;
  const int hi = lane >> 4, ln = lane & 15;

  v8f acc[2][4];
#pragma unroll
  for (int i = 0; i < 2; ++i)
#pragma unroll
    for (int j = 0; j < 4; ++j) acc[i][j] = zero8();

  for (int k0 = 0; k0 < EMBED; k0 += 32) {
    __syncthreads();
#pragma unroll
    for (int i = 0; i < 2; ++i) {
      int u = t + i * 256;
      int row = u >> 2, seg = u & 3;
      cp16_to_lds(As + row * TS + seg * 8,
                  A + (size_t)(bm + row) * EMBED + k0 + seg * 8);
      cp16_to_lds(Bs + row * TS + seg * 8,
                  Wob + (size_t)(bn + row) * EMBED + k0 + seg * 8);
    }
    wait_lds_async();
    __syncthreads();

    v16bf af[2], bfr[4];
#pragma unroll
    for (int mi = 0; mi < 2; ++mi) af[mi] = frag_a(As + (mw * 32 + mi * 16) * TS, TS);
#pragma unroll
    for (int ni = 0; ni < 4; ++ni) bfr[ni] = frag_b_nk(Bs + (nw * 64 + ni * 16) * TS, TS);
#pragma unroll
    for (int mi = 0; mi < 2; ++mi)
#pragma unroll
      for (int ni = 0; ni < 4; ++ni)
        acc[mi][ni] = wmma_bf16(af[mi], bfr[ni], acc[mi][ni]);
  }

#pragma unroll
  for (int mi = 0; mi < 2; ++mi)
#pragma unroll
    for (int ni = 0; ni < 4; ++ni)
#pragma unroll
      for (int r = 0; r < 8; ++r) {
        int m = bm + mw * 32 + mi * 16 + r + hi * 8;
        int f = bn + nw * 64 + ni * 16 + ln;
        Out[(size_t)m * EMBED + f] = acc[mi][ni][r] + bo[f];
      }
}

extern "C" void kernel_launch(void* const* d_in, const int* in_sizes, int n_in,
                              void* d_out, int out_size, void* d_ws, size_t ws_size,
                              hipStream_t stream) {
  const float* hs = (const float*)d_in[0];
  const float* Wq = (const float*)d_in[1];
  const float* bq = (const float*)d_in[2];
  const float* Wk = (const float*)d_in[3];
  const float* bk = (const float*)d_in[4];
  const float* Wv = (const float*)d_in[5];
  const float* bv = (const float*)d_in[6];
  const float* Wo = (const float*)d_in[7];
  const float* bo = (const float*)d_in[8];
  const int* act = (const int*)d_in[9];
  float* out = (float*)d_out;

  char* base = (char*)d_ws;
  size_t off = 0;
  auto carve = [&](size_t bytes) -> char* {
    char* r = base + off;
    off += (bytes + 255) & ~(size_t)255;
    return r;
  };
  __bf16* Xb  = (__bf16*)carve((size_t)MROWS * EMBED * 2);   // 16 MB
  __bf16* Wqb = (__bf16*)carve((size_t)EMBED * EMBED * 2);   //  2 MB (native [f,e])
  __bf16* Wkb = (__bf16*)carve((size_t)EMBED * EMBED * 2);
  __bf16* Wvb = (__bf16*)carve((size_t)EMBED * EMBED * 2);
  __bf16* Wob = (__bf16*)carve((size_t)EMBED * EMBED * 2);
  __bf16* Qb  = (__bf16*)carve((size_t)MROWS * EMBED * 2);   // 16 MB
  __bf16* Kb  = (__bf16*)carve((size_t)MROWS * EMBED * 2);
  __bf16* Vb  = (__bf16*)carve((size_t)MROWS * EMBED * 2);
  __bf16* Ab  = (__bf16*)carve((size_t)MROWS * EMBED * 2);   // attn (pre-proj)
  (void)ws_size; (void)in_sizes; (void)n_in; (void)out_size;

  const int WN = EMBED * EMBED;
  cvt_f32_bf16<<<(MROWS * EMBED) / 1024, 256, 0, stream>>>(hs, Xb, MROWS * EMBED);
  cvt_f32_bf16<<<WN / 1024, 256, 0, stream>>>(Wq, Wqb, WN);
  cvt_f32_bf16<<<WN / 1024, 256, 0, stream>>>(Wk, Wkb, WN);
  cvt_f32_bf16<<<WN / 1024, 256, 0, stream>>>(Wv, Wvb, WN);
  cvt_f32_bf16<<<WN / 1024, 256, 0, stream>>>(Wo, Wob, WN);
  qkv_gemm<<<dim3(MROWS / 128, EMBED / 128, 3), 256, 0, stream>>>(
      Xb, Wqb, Wkb, Wvb, bq, bk, bv, Qb, Kb, Vb);
  flash_attn<<<dim3(SEQ / 128, NHEADS, BATCH), 256, 0, stream>>>(Qb, Kb, Vb, Ab, act);
  out_gemm<<<dim3(MROWS / 128, EMBED / 128), 256, 0, stream>>>(Ab, Wob, bo, out);
}
